// TwoViewAttentionModule_91293824844120
// MI455X (gfx1250) — matrix operand order repaired and checked by reading, not compiled, not run on hardware
//
#include <hip/hip_runtime.h>
#include <hip/hip_bf16.h>

typedef __attribute__((ext_vector_type(16))) __bf16 v16bf;
typedef __attribute__((ext_vector_type(8)))  float  v8f;

#define HEADS 8
#define EMBED 64
#define FEAT  256
#define NSEQ  2048
#define BATCH 2

__device__ __forceinline__ v8f wmma_bf16(v16bf a, v16bf b, v8f c) {
  return __builtin_amdgcn_wmma_f32_16x16x32_bf16(false, a, false, b, (short)0, c, false, false);
}

// low 32 bits of a generic pointer to a __shared__ object == LDS byte offset
__device__ __forceinline__ unsigned int lds_off(const void* p) {
  return (unsigned int)(unsigned long long)p;
}

// async DMA: 16 bytes global -> LDS, tracked by ASYNCcnt (ISA 10.x / 15.18.3)
__device__ __forceinline__ void async_g2l_b128(unsigned int ldst, const void* gsrc) {
  asm volatile("global_load_async_to_lds_b128 %0, %1, off"
               :: "v"(ldst), "v"(gsrc) : "memory");
}

// ---------------------------------------------------------------------------
// Kernel 0: zero the two l1 output slots (d_out is poisoned by the harness).
// ---------------------------------------------------------------------------
__global__ void zero_l1_kernel(float* l1out) {
  if (threadIdx.x < BATCH) l1out[threadIdx.x] = 0.0f;
}

// ---------------------------------------------------------------------------
// Kernel 0b: pre-convert z_b (v matrix) to bf16 so the attention hot loop can
// async-DMA it straight into LDS with no per-iteration VALU conversion.
// ---------------------------------------------------------------------------
__global__ __launch_bounds__(256) void vpack_kernel(const float* __restrict__ z,
                                                    __bf16* __restrict__ vb) {
  const size_t i = ((size_t)blockIdx.x * 256 + threadIdx.x) * 4;
#pragma unroll
  for (int j = 0; j < 4; ++j) vb[i + j] = (__bf16)z[i + j];
}

// ---------------------------------------------------------------------------
// Kernel 1: embeddings.  q = (W_a @ z_a) * 1/sqrt(E),  k = W_b @ z_b.
// Block = 8 waves sharing one (b, mat, n-tile); wave w owns o-tile og*8+w.
// z tile [256f x 16n] staged once into LDS -> B fragment = one 32B ds_load.
// Output layout: qT/kT[b][h][n][e] bf16 -> attention fragments contiguous.
// ---------------------------------------------------------------------------
__global__ __launch_bounds__(256) void embed_kernel(
    const float* __restrict__ z_a, const float* __restrict__ z_b,
    const float* __restrict__ W_a, const float* __restrict__ W_b,
    __bf16* __restrict__ qT, __bf16* __restrict__ kT)
{
  __shared__ __bf16 Zlds[16][272];              // [n][f], 544B row (32B-aligned)

  const int wave = threadIdx.x >> 5;
  const int lane = threadIdx.x & 31;
  const int half = lane >> 4;
  const int ln   = lane & 15;

  const int blk = blockIdx.x;                   // 2048 blocks
  const int nt  = blk & 127;                    // 2048/16 n-tiles
  const int og  = (blk >> 7) & 3;               // 4 groups of 8 o-tiles
  const int mat = (blk >> 9) & 1;               // 0 = q, 1 = k
  const int b   = blk >> 10;

  const float* W = mat ? W_b : W_a;             // [512, 256] row-major
  const float* Z = mat ? z_b : z_a;             // [B, 256, 2048]
  __bf16* dst    = mat ? kT  : qT;
  const float scale = mat ? 1.0f : 0.125f;      // fold 1/sqrt(64) into q

  const int n0 = nt * 16;
  const int o0 = (og * 8 + wave) * 16;

  // ---- stage z tile: thread f reads 16 contiguous floats, writes column f
  {
    const int f = threadIdx.x;
    const float* zp = Z + ((size_t)b * FEAT + f) * NSEQ + n0;
#pragma unroll
    for (int n = 0; n < 16; ++n) Zlds[n][f] = (__bf16)zp[n];
  }
  __syncthreads();

  v8f acc = {};
  for (int ks = 0; ks < 8; ++ks) {
    const int f0 = ks * 32;
    v16bf a;
#pragma unroll
    for (int e = 0; e < 16; ++e) {
      // A (16x32): M = lane%16, K = half*16 + e
      const int Ka = f0 + (half << 4) + e;
      a[e] = (__bf16)W[(size_t)(o0 + ln) * FEAT + Ka];
    }
    // B (32x16): K = half*16 + e, N = lane%16 -> one aligned v16bf LDS load
    v16bf bb = *(const v16bf*)&Zlds[ln][f0 + half * 16];
    acc = wmma_bf16(a, bb, acc);
  }
  // C layout: row M = r + half*8, col N = ln.  Store transposed as [n][e].
#pragma unroll
  for (int r = 0; r < 8; ++r) {
    const int o = o0 + r + half * 8;            // 0..511
    const int h = o >> 6, e = o & 63;
    dst[(((size_t)b * HEADS + h) * NSEQ + (n0 + ln)) * EMBED + e] =
        (__bf16)(acc[r] * scale);
  }
}

// ---------------------------------------------------------------------------
// Kernel 2: fused flash attention + L1 + head-combine.
// Block = 256 threads = 8 waves; block owns (batch b, 16 queries).
// Wave h runs online-softmax attention for head h over all 2048 keys,
// accumulating U_h[16x256] in 16 f32 WMMA accumulators.  The shared V tile
// is double-buffered in LDS and filled one iteration ahead with
// global_load_async_to_lds_b128 (ASYNCcnt).  Finally all waves cooperatively
// compute out[16x256] = sum_h U_h @ w_h through LDS.
// ---------------------------------------------------------------------------
__global__ __launch_bounds__(256) void attn_kernel(
    const __bf16* __restrict__ qT, const __bf16* __restrict__ kT,
    const __bf16* __restrict__ vb16, const float* __restrict__ W_c,
    float* __restrict__ out, float* __restrict__ l1out)
{
  __shared__ __bf16 Plds[8][16][40];     // P tile per wave (C->A relayout)
  __shared__ __bf16 Vlds[2][256][48];    // V tile [f][k], 96B row, dbl-buffered
  __shared__ __bf16 Ulds[8][16][264];    // normalized U per head

  const int h    = threadIdx.x >> 5;     // wave id == head
  const int lane = threadIdx.x & 31;
  const int half = lane >> 4;
  const int ln   = lane & 15;
  const int b    = blockIdx.x >> 7;
  const int q0   = (blockIdx.x & 127) * 16;
  const size_t bh = (size_t)b * HEADS + h;

  // Q A-fragments for e=0..31 and e=32..63 (held for the whole key loop)
  const __bf16* qrow = qT + (bh * NSEQ + q0 + ln) * EMBED;
  v16bf Qa0, Qa1;
#pragma unroll
  for (int e = 0; e < 16; ++e) {
    const int K = (half << 4) + e;
    Qa0[e] = qrow[K];
    Qa1[e] = qrow[32 + K];
  }

  v8f U[16];
#pragma unroll
  for (int t = 0; t < 16; ++t) U[t] = (v8f){};
  float m[8], l[8];
#pragma unroll
  for (int r = 0; r < 8; ++r) { m[r] = -3.0e38f; l[r] = 0.0f; }
  float l1acc = 0.0f;

  const __bf16* kbase = kT + bh * NSEQ * EMBED;
  const __bf16* vbase = vb16 + ((size_t)b * FEAT + threadIdx.x) * NSEQ;

  // ---- prime V buffer 0: each thread DMAs its 64B row slice (4 x b128)
#pragma unroll
  for (int c = 0; c < 4; ++c)
    async_g2l_b128(lds_off(&Vlds[0][threadIdx.x][c * 8]), vbase + c * 8);

  for (int it = 0; it < 64; ++it) {
    const int k0  = it * 32;
    const int cur = it & 1;

    // ---- prefetch next iteration's K rows (global_prefetch_b8)
    if (it + 1 < 64) {
      const __bf16* nk = kbase + (size_t)(k0 + 32 + ln) * EMBED;
      __builtin_prefetch(nk, 0, 1);
      __builtin_prefetch(nk + (size_t)16 * EMBED, 0, 1);
    }

    // ---- S = q^T k for 16 queries x 32 keys (two 16x16 tiles) ----
    v8f S0 = {}, S1 = {};
    {
      v16bf kb;
      kb = *(const v16bf*)(kbase + (size_t)(k0 + ln) * EMBED + half * 16);
      S0 = wmma_bf16(Qa0, kb, S0);
      kb = *(const v16bf*)(kbase + (size_t)(k0 + ln) * EMBED + 32 + half * 16);
      S0 = wmma_bf16(Qa1, kb, S0);
      kb = *(const v16bf*)(kbase + (size_t)(k0 + 16 + ln) * EMBED + half * 16);
      S1 = wmma_bf16(Qa0, kb, S1);
      kb = *(const v16bf*)(kbase + (size_t)(k0 + 16 + ln) * EMBED + 32 + half * 16);
      S1 = wmma_bf16(Qa1, kb, S1);
    }
#pragma unroll
    for (int r = 0; r < 8; ++r) l1acc += fabsf(S0[r]) + fabsf(S1[r]);

    // ---- online softmax (rows live across the 16 lanes of each half) ----
    float ps[8];
#pragma unroll
    for (int r = 0; r < 8; ++r) {
      float v = fmaxf(S0[r], S1[r]);
#pragma unroll
      for (int msk = 1; msk < 16; msk <<= 1) v = fmaxf(v, __shfl_xor(v, msk, 32));
      const float mn = fmaxf(m[r], v);
      ps[r] = __expf(m[r] - mn);
      m[r]  = mn;
      const float p0 = __expf(S0[r] - mn);
      const float p1 = __expf(S1[r] - mn);
      S0[r] = p0; S1[r] = p1;
      float s = p0 + p1;
#pragma unroll
      for (int msk = 1; msk < 16; msk <<= 1) s += __shfl_xor(s, msk, 32);
      l[r] = l[r] * ps[r] + s;
    }
#pragma unroll
    for (int t = 0; t < 16; ++t)
#pragma unroll
      for (int r = 0; r < 8; ++r) U[t][r] *= ps[r];

    __syncthreads();  // closes prior iteration's reads of Plds / Vlds[1-cur]

    // ---- stage P tile; DMA next V tile into the other buffer ----
#pragma unroll
    for (int r = 0; r < 8; ++r) {
      Plds[h][r + half * 8][ln]      = (__bf16)S0[r];
      Plds[h][r + half * 8][ln + 16] = (__bf16)S1[r];
    }
    if (it + 1 < 64) {
      const __bf16* vp = vbase + (k0 + 32);
#pragma unroll
      for (int c = 0; c < 4; ++c)
        async_g2l_b128(lds_off(&Vlds[1 - cur][threadIdx.x][c * 8]), vp + c * 8);
      // current buffer's 4 copies (issued last iteration) must have landed
      asm volatile("s_wait_asynccnt 0x4" ::: "memory");
    } else {
      asm volatile("s_wait_asynccnt 0x0" ::: "memory");
    }
    __syncthreads();

    // ---- U += P @ V^T  (K = 32 keys, 16 column tiles) ----
    v16bf Pa;
#pragma unroll
    for (int e = 0; e < 16; ++e) {
      const int K = (half << 4) + e;
      Pa[e] = Plds[h][ln][K];
    }
#pragma unroll
    for (int t = 0; t < 16; ++t) {
      v16bf vb = *(const v16bf*)&Vlds[cur][t * 16 + ln][half * 16];
      U[t] = wmma_bf16(Pa, vb, U[t]);
    }
  }

  // ---- L1: wave reduce, one atomic per wave ----
  float tot = l1acc;
#pragma unroll
  for (int msk = 1; msk < 32; msk <<= 1) tot += __shfl_xor(tot, msk, 32);
  if (lane == 0)
    atomicAdd(&l1out[b], tot * (1.0f / ((float)HEADS * NSEQ * NSEQ)));

  // ---- normalize U rows, stage to LDS ----
#pragma unroll
  for (int r = 0; r < 8; ++r) l[r] = 1.0f / l[r];
#pragma unroll
  for (int t = 0; t < 16; ++t)
#pragma unroll
    for (int r = 0; r < 8; ++r)
      Ulds[h][r + half * 8][t * 16 + ln] = (__bf16)(U[t][r] * l[r]);
  __syncthreads();

  // ---- out[q,o] = sum_h sum_f U[h][q][f] * W_c[o][h*256+f] ----
#pragma unroll
  for (int oi = 0; oi < 2; ++oi) {
    const int o0 = (h + oi * 8) * 16;          // wave owns 2 of 16 col tiles
    v8f acc = {};
    for (int h2 = 0; h2 < 8; ++h2) {
#pragma unroll
      for (int ks = 0; ks < 8; ++ks) {
        const int f0 = ks * 32;
        v16bf a, bb;
#pragma unroll
        for (int e = 0; e < 16; ++e) {
          const int Ka = f0 + (half << 4) + e;
          a[e] = Ulds[h2][ln][Ka];
          const int Kb = f0 + (half << 4) + e;  // contiguous in W_c row
          bb[e] = (__bf16)W_c[(size_t)(o0 + ln) * (HEADS * FEAT) + h2 * FEAT + Kb];
        }
        acc = wmma_bf16(a, bb, acc);
      }
    }
#pragma unroll
    for (int r = 0; r < 8; ++r)
      out[((size_t)b * NSEQ + q0 + r + half * 8) * FEAT + o0 + ln] = acc[r];
  }
}

// ---------------------------------------------------------------------------
extern "C" void kernel_launch(void* const* d_in, const int* in_sizes, int n_in,
                              void* d_out, int out_size, void* d_ws, size_t ws_size,
                              hipStream_t stream) {
  (void)in_sizes; (void)n_in; (void)out_size; (void)ws_size;
  const float* z_a = (const float*)d_in[0];   // [2, 256, 2048]
  const float* z_b = (const float*)d_in[1];   // [2, 256, 2048]
  const float* W_a = (const float*)d_in[2];   // [512, 256]
  const float* W_b = (const float*)d_in[3];   // [512, 256]
  const float* W_c = (const float*)d_in[4];   // [256, 2048]

  float* out   = (float*)d_out;                       // [2, 2048, 256]
  float* l1out = out + (size_t)BATCH * NSEQ * FEAT;   // [2]

  __bf16* qT   = (__bf16*)d_ws;                                  // 4 MB
  __bf16* kT   = qT + (size_t)BATCH * HEADS * NSEQ * EMBED;      // 4 MB
  __bf16* vb16 = kT + (size_t)BATCH * HEADS * NSEQ * EMBED;      // 2 MB

  zero_l1_kernel<<<1, 32, 0, stream>>>(l1out);
  vpack_kernel<<<(BATCH * FEAT * NSEQ) / 1024, 256, 0, stream>>>(z_b, vb16);
  embed_kernel<<<2048, 256, 0, stream>>>(z_a, z_b, W_a, W_b, qT, kT);
  attn_kernel<<<BATCH * (NSEQ / 16), 256, 0, stream>>>(qT, kT, vb16, W_c, out, l1out);
}
